// ScaledDotProductAttention_73212012528233
// MI455X (gfx1250) — matrix operand order repaired and checked by reading, not compiled, not run on hardware
//
#include <hip/hip_runtime.h>

typedef __attribute__((ext_vector_type(16))) _Float16 v16h;
typedef __attribute__((ext_vector_type(8)))  float    v8f;
typedef __attribute__((ext_vector_type(4)))  float    v4f;
typedef __attribute__((ext_vector_type(4)))  int      v4i;

#define S_LEN 1024
#define D_DIM 64
#define NEGV  (-1.0e9f)

__device__ __forceinline__ float wave_max(float v) {
#pragma unroll
  for (int off = 16; off > 0; off >>= 1)
    v = fmaxf(v, __shfl_xor(v, off, 32));
  return v;
}
__device__ __forceinline__ float wave_sum(float v) {
#pragma unroll
  for (int off = 16; off > 0; off >>= 1)
    v += __shfl_xor(v, off, 32);
  return v;
}

__global__ __launch_bounds__(128)
void fused_graph_attn(const float* __restrict__ Q,
                      const float* __restrict__ K,
                      const float* __restrict__ V,
                      const int*   __restrict__ mask,
                      const float* __restrict__ adj,
                      const float* __restrict__ dist,
                      const float* __restrict__ cw,
                      const float* __restrict__ cb,
                      float* __restrict__ ctx_out,
                      float* __restrict__ attn_out) {
  // 4 waves per block, each wave owns a 16-row score slab: 4*16*1024*4B = 256 KB
  __shared__ float lds[4][16][S_LEN];

  const int lane = threadIdx.x & 31;
  const int wid  = threadIdx.x >> 5;
  const int bh   = blockIdx.x >> 4;     // 0..31  (B*H)
  const int qt   = blockIdx.x & 15;     // 0..15  (S / 64)
  const int q0   = qt * 64 + wid * 16;  // first q row of this wave

  const int m16 = lane & 15;
  const int hi  = lane >> 4;

  const float scale = 0.125f;           // 1/sqrt(64)
  const float w0 = cw[0], w1 = cw[1], w2 = cw[2], cbias = cb[0];

  const size_t qkv_base = (size_t)bh * S_LEN * D_DIM;
  const size_t mat_base = (size_t)bh * S_LEN * S_LEN;

  // ---------------- Phase 1: scores = Q * K^T  -> LDS ----------------
  // 16-bit A(16x32) layout: lane m = l&15, hi = l>>4,
  //   halves 2v,2v+1 hold k = (v>>2)*16 + hi*8 + (v&3)*2 + {0,1}
  v16h aq0, aq1;
  {
    const float* qrow = Q + qkv_base + (size_t)(q0 + m16) * D_DIM;
#pragma unroll
    for (int v = 0; v < 8; ++v) {
      const int kb = (v >> 2) * 16 + hi * 8 + (v & 3) * 2;
      aq0[2 * v]     = (_Float16)qrow[kb];
      aq0[2 * v + 1] = (_Float16)qrow[kb + 1];
      aq1[2 * v]     = (_Float16)qrow[32 + kb];
      aq1[2 * v + 1] = (_Float16)qrow[32 + kb + 1];
    }
  }

  for (int kt = 0; kt < 64; ++kt) {
    // B fragments: B[kk][n] = K[kt*16 + n][kk]
    // 16-bit B(32x16) layout: lane n = l&15, halves 2v,2v+1 hold kk = (v>>2)*16 + hi*8 + (v&3)*2 + {0,1}
    const float* krow = K + qkv_base + (size_t)(kt * 16 + m16) * D_DIM;
    v16h bk0, bk1;
#pragma unroll
    for (int v = 0; v < 8; ++v) {
      const int kb = (v >> 2) * 16 + hi * 8 + (v & 3) * 2;
      bk0[2 * v]     = (_Float16)krow[kb];
      bk0[2 * v + 1] = (_Float16)krow[kb + 1];
      bk1[2 * v]     = (_Float16)krow[32 + kb];
      bk1[2 * v + 1] = (_Float16)krow[32 + kb + 1];
    }
    v8f c = {};
    c = __builtin_amdgcn_wmma_f32_16x16x32_f16(false, aq0, false, bk0,
                                               (short)0, c, false, false);
    c = __builtin_amdgcn_wmma_f32_16x16x32_f16(false, aq1, false, bk1,
                                               (short)0, c, false, false);
    // C(16x16) f32 layout: VGPR r -> row r + 8*hi, col = l&15
#pragma unroll
    for (int r = 0; r < 8; ++r)
      lds[wid][r + 8 * hi][kt * 16 + m16] = c[r] * scale;
  }

  // -------- Phase 2: double softmax + conv mix, row by row --------
  // float4-vectorized sweeps; stream-once tensors (mask/dist/adj/attn) use
  // non-temporal hints so the L2 keeps the heavily re-gathered Q/K/V slices.
  for (int m = 0; m < 16; ++m) {
    const size_t rbase = mat_base + (size_t)(q0 + m) * S_LEN;  // 16B-aligned
    float*       row  = lds[wid][m];
    v4f*         rowv = (v4f*)row;
    const v4i*   mv   = (const v4i*)(mask + rbase);
    const v4f*   dv   = (const v4f*)(dist + rbase);
    const v4f*   av   = (const v4f*)(adj  + rbase);
    v4f*         ov   = (v4f*)      (attn_out + rbase);

    // sweep 1: apply mask (read ONCE, NT), find max1
    float mx = -3.0e38f;
#pragma unroll 2
    for (int j = lane; j < S_LEN / 4; j += 32) {
      v4f s = rowv[j];
      const v4i mk = __builtin_nontemporal_load(&mv[j]);
#pragma unroll
      for (int e = 0; e < 4; ++e) {
        s[e] = mk[e] ? NEGV : s[e];
        mx = fmaxf(mx, s[e]);
      }
      rowv[j] = s;
    }
    mx = wave_max(mx);

    // sweep 2: softmax-1 denominator
    float sm = 0.0f;
#pragma unroll 2
    for (int j = lane; j < S_LEN / 4; j += 32) {
      const v4f s = rowv[j];
#pragma unroll
      for (int e = 0; e < 4; ++e)
        sm += __expf(s[e] - mx);
    }
    sm = wave_sum(sm);
    const float inv1 = 1.0f / sm;

    // sweep 3: weighted = w0*p1 + w1*dist + w2*adj + b (NT reads), re-mask, max2
    float mx2 = -3.0e38f;
#pragma unroll 2
    for (int j = lane; j < S_LEN / 4; j += 32) {
      const v4f s = rowv[j];
      const v4f d = __builtin_nontemporal_load(&dv[j]);
      const v4f a = __builtin_nontemporal_load(&av[j]);
      v4f w;
#pragma unroll
      for (int e = 0; e < 4; ++e) {
        const float p1 = __expf(s[e] - mx) * inv1;
        float we = w0 * p1 + w1 * d[e] + w2 * a[e] + cbias;
        we = (s[e] < -0.5e9f) ? NEGV : we;
        w[e] = we;
        mx2 = fmaxf(mx2, we);
      }
      rowv[j] = w;
    }
    mx2 = wave_max(mx2);

    // sweep 4: softmax-2 exponentials + denominator
    float sm2 = 0.0f;
#pragma unroll 2
    for (int j = lane; j < S_LEN / 4; j += 32) {
      v4f w = rowv[j];
#pragma unroll
      for (int e = 0; e < 4; ++e) {
        w[e] = __expf(w[e] - mx2);
        sm2 += w[e];
      }
      rowv[j] = w;
    }
    sm2 = wave_sum(sm2);
    const float inv2 = 1.0f / sm2;

    // sweep 5: final attn -> LDS (for GEMM2) and global (NT b128 store)
#pragma unroll 2
    for (int j = lane; j < S_LEN / 4; j += 32) {
      v4f e4 = rowv[j];
#pragma unroll
      for (int e = 0; e < 4; ++e)
        e4[e] *= inv2;
      rowv[j] = e4;
      __builtin_nontemporal_store(e4, &ov[j]);
    }
  }

  __syncthreads();

  // ---------------- Phase 3: context = attn * V ----------------
  v8f acc0 = {}, acc1 = {}, acc2 = {}, acc3 = {};
  for (int kt = 0; kt < 32; ++kt) {  // 1024 keys in K=32 chunks
    // A fragment from LDS attn slab
    v16h aa;
#pragma unroll
    for (int v = 0; v < 8; ++v) {
      const int kb = (v >> 2) * 16 + hi * 8 + (v & 3) * 2;
      aa[2 * v]     = (_Float16)lds[wid][m16][kt * 32 + kb];
      aa[2 * v + 1] = (_Float16)lds[wid][m16][kt * 32 + kb + 1];
    }
    // 4 N-tiles of 16 cover D=64; B[kk][n] = V[kt*32 + kk][nt*16 + n]
#pragma unroll
    for (int nt = 0; nt < 4; ++nt) {
      v16h bb;
#pragma unroll
      for (int v = 0; v < 8; ++v) {
        const int kk = kt * 32 + (v >> 2) * 16 + hi * 8 + (v & 3) * 2;
        const float* vp = V + qkv_base + (size_t)kk * D_DIM + nt * 16 + m16;
        bb[2 * v]     = (_Float16)vp[0];
        bb[2 * v + 1] = (_Float16)vp[D_DIM];
      }
      if      (nt == 0) acc0 = __builtin_amdgcn_wmma_f32_16x16x32_f16(false, aa, false, bb, (short)0, acc0, false, false);
      else if (nt == 1) acc1 = __builtin_amdgcn_wmma_f32_16x16x32_f16(false, aa, false, bb, (short)0, acc1, false, false);
      else if (nt == 2) acc2 = __builtin_amdgcn_wmma_f32_16x16x32_f16(false, aa, false, bb, (short)0, acc2, false, false);
      else              acc3 = __builtin_amdgcn_wmma_f32_16x16x32_f16(false, aa, false, bb, (short)0, acc3, false, false);
    }
  }

  // store context (8 MB total, minor traffic)
#pragma unroll
  for (int r = 0; r < 8; ++r) {
    const size_t ro = qkv_base + (size_t)(q0 + r + 8 * hi) * D_DIM + m16;
    ctx_out[ro + 0]  = acc0[r];
    ctx_out[ro + 16] = acc1[r];
    ctx_out[ro + 32] = acc2[r];
    ctx_out[ro + 48] = acc3[r];
  }
}

extern "C" void kernel_launch(void* const* d_in, const int* in_sizes, int n_in,
                              void* d_out, int out_size, void* d_ws, size_t ws_size,
                              hipStream_t stream) {
  (void)in_sizes; (void)n_in; (void)d_ws; (void)ws_size; (void)out_size;
  const float* Q    = (const float*)d_in[0];
  const float* K    = (const float*)d_in[1];
  const float* V    = (const float*)d_in[2];
  const int*   mask = (const int*)d_in[3];
  const float* adj  = (const float*)d_in[4];
  const float* dist = (const float*)d_in[5];
  const float* cw   = (const float*)d_in[6];
  const float* cb   = (const float*)d_in[7];

  float* ctx  = (float*)d_out;                         // [4,8,1024,64]
  float* attn = ctx + (size_t)4 * 8 * 1024 * 64;       // [4,8,1024,1024]

  dim3 grid(512);   // (B*H) * (S/64) workgroups
  dim3 block(128);  // 4 waves, 16 q-rows each
  hipLaunchKernelGGL(fused_graph_attn, grid, block, 0, stream,
                     Q, K, V, mask, adj, dist, cw, cb, ctx, attn);
}
